// SpikeMLP_88098369175788
// MI455X (gfx1250) — matrix operand order repaired and checked by reading, not compiled, not run on hardware
//
#include <hip/hip_runtime.h>
#include <hip/hip_bf16.h>

typedef __attribute__((ext_vector_type(16))) _Float16 v16h;
typedef __attribute__((ext_vector_type(8)))  float    v8f;
typedef __attribute__((ext_vector_type(4)))  float    f32x4;

static constexpr float    kDT    = 0.02f;
static constexpr int      kNumT  = 151;   // t = 0..150
static constexpr unsigned kLastT = 150u;  // forced spike index

// ---------------------------------------------------------------------------
// Prep kernels
// ---------------------------------------------------------------------------

// Pad ti rows from Kreal -> Kpad; pad entries get a huge spike time so
// relu(t - ti) == 0 for padded columns.
__global__ void pad_ti_kernel(const float* __restrict__ ti, float* __restrict__ tiP,
                              int Kreal, int Kpad, int total) {
  int i = blockIdx.x * blockDim.x + threadIdx.x;
  if (i >= total) return;
  int r = i / Kpad;
  int c = i - r * Kpad;
  tiP[i] = (c < Kreal) ? ti[r * Kreal + c] : 1.0e9f;
}

// Pack W[Fout, Kreal] (row-major f32) into f16 WMMA B-operand tiles.
// Tile (kt, nt) covers K rows kt*32..+31, N cols nt*16..+15.
// Per-lane layout (16-bit B 32x16): lane<16 -> col n=lane,    K = kt*32 + {0..7,16..23}[e];
//                                   lane>=16 -> col n=lane-16, K = kt*32 + {8..15,24..31}[e].
// Flat index: (((kt*nTiles + nt)*32) + lane)*16 + e.
__global__ void pack_w_kernel(const float* __restrict__ W, _Float16* __restrict__ Bp,
                              int Kreal, int Wrows, int nTiles, int total) {
  int e = blockIdx.x * blockDim.x + threadIdx.x;
  if (e >= total) return;
  int tile   = e >> 9;
  int within = e & 511;
  int lane   = within >> 4;
  int e16    = within & 15;
  int kt = tile / nTiles;
  int nt = tile - kt * nTiles;
  int k = kt * 32 + ((lane & 16) ? 8 : 0) + (e16 & 7) + ((e16 & 8) ? 16 : 0);
  int n = nt * 16 + (lane & 15);
  float v = (n < Wrows && k < Kreal) ? W[n * Kreal + k] : 0.0f;
  Bp[e] = (_Float16)v;
}

__global__ void fill_u32_kernel(unsigned* __restrict__ p, unsigned v, int n) {
  int i = blockIdx.x * blockDim.x + threadIdx.x;
  if (i < n) p[i] = v;
}

// h1[b,j] = idx1[b,j] * dt  -> becomes layer-2 spike-time input
__global__ void idx_to_time_kernel(const unsigned* __restrict__ idx,
                                   float* __restrict__ t, int n) {
  int i = blockIdx.x * blockDim.x + threadIdx.x;
  if (i < n) t[i] = (float)idx[i] * kDT;
}

// out[b*10 + j] = idx2[b*16 + j] * dt
__global__ void final_out_kernel(const unsigned* __restrict__ idx2,
                                 float* __restrict__ out, int total) {
  int i = blockIdx.x * blockDim.x + threadIdx.x;
  if (i >= total) return;
  int b = i / 10;
  int j = i - 10 * b;
  out[i] = (float)idx2[b * 16 + j] * kDT;
}

// ---------------------------------------------------------------------------
// Fused spike-GEMM with one-K-step software pipelining.
// One wave computes a 16-row (b,t) x NT*16-col tile of
// mem = relu(t - ti) @ W^T via v_wmma_f32_16x16x32_f16, building the A
// fragment on the fly from ti, then atomicMin-ing first-spike indices.
// ---------------------------------------------------------------------------
template <int NT>
__global__ void spike_gemm_kernel(const float* __restrict__ tiPad, int Kpad,
                                  const _Float16* __restrict__ Bpack, int nTilesTotal,
                                  unsigned* __restrict__ idx, int idxStride,
                                  int nGroups, int totalWaves) {
  int gtid = blockIdx.x * blockDim.x + threadIdx.x;
  int wave = gtid >> 5;
  int lane = gtid & 31;
  if (wave >= totalWaves) return;   // wave-uniform: EXEC stays all-1 for WMMA

  int mTile  = wave / nGroups;
  int ng     = wave - mTile * nGroups;
  int ngBase = ng * NT;

  // A-operand addressing: lanes 0-15 and 16-31 both map to rows 0..15,
  // covering K sub-blocks {0..7,16..23} and {8..15,24..31} respectively.
  int laneHi = (lane >> 4) & 1;
  int rowA   = mTile * 16 + (lane & 15);
  int bA     = rowA / kNumT;
  int tA     = rowA - bA * kNumT;
  float tval = (float)tA * kDT;

  const float*    tPtr = tiPad + (size_t)bA * Kpad + (laneHi << 3);
  const _Float16* bPtr = Bpack + ((size_t)ngBase * 32 + lane) * 16;
  const size_t    bStep = (size_t)nTilesTotal * 512;  // halves per K-step

  v8f acc[NT];
#pragma unroll
  for (int it = 0; it < NT; ++it)
#pragma unroll
    for (int r = 0; r < 8; ++r) acc[it][r] = 0.0f;

  // ---- prologue: load K-step 0 operand data -------------------------------
  f32x4 f0 = *(const f32x4*)(tPtr);
  f32x4 f1 = *(const f32x4*)(tPtr + 4);
  f32x4 f2 = *(const f32x4*)(tPtr + 16);
  f32x4 f3 = *(const f32x4*)(tPtr + 20);
  v16h bf[NT];
#pragma unroll
  for (int it = 0; it < NT; ++it)
    bf[it] = *(const v16h*)(bPtr + (size_t)it * 512);  // it*1024B imm offsets

  int Ksteps = Kpad >> 5;
  for (int kt = 0; kt < Ksteps - 1; ++kt) {
    tPtr += 32;
    bPtr += bStep;
    // ---- prefetch next K-step into fresh registers (overlaps the WMMAs) ---
    f32x4 nf0 = *(const f32x4*)(tPtr);
    f32x4 nf1 = *(const f32x4*)(tPtr + 4);
    f32x4 nf2 = *(const f32x4*)(tPtr + 16);
    f32x4 nf3 = *(const f32x4*)(tPtr + 20);
    v16h nbf[NT];
#pragma unroll
    for (int it = 0; it < NT; ++it)
      nbf[it] = *(const v16h*)(bPtr + (size_t)it * 512);

    // ---- build A fragment for current step --------------------------------
    v16h a;
#pragma unroll
    for (int e = 0; e < 4; ++e) {
      a[e]      = (_Float16)fmaxf(tval - f0[e], 0.0f);
      a[e + 4]  = (_Float16)fmaxf(tval - f1[e], 0.0f);
      a[e + 8]  = (_Float16)fmaxf(tval - f2[e], 0.0f);
      a[e + 12] = (_Float16)fmaxf(tval - f3[e], 0.0f);
    }
    // ---- matrix ops --------------------------------------------------------
#pragma unroll
    for (int it = 0; it < NT; ++it)
      acc[it] = __builtin_amdgcn_wmma_f32_16x16x32_f16(
          false, a, false, bf[it], (short)0, acc[it], false, false);

    // rotate double buffers
    f0 = nf0; f1 = nf1; f2 = nf2; f3 = nf3;
#pragma unroll
    for (int it = 0; it < NT; ++it) bf[it] = nbf[it];
  }

  // ---- epilogue K-step (no prefetch) --------------------------------------
  {
    v16h a;
#pragma unroll
    for (int e = 0; e < 4; ++e) {
      a[e]      = (_Float16)fmaxf(tval - f0[e], 0.0f);
      a[e + 4]  = (_Float16)fmaxf(tval - f1[e], 0.0f);
      a[e + 8]  = (_Float16)fmaxf(tval - f2[e], 0.0f);
      a[e + 12] = (_Float16)fmaxf(tval - f3[e], 0.0f);
    }
#pragma unroll
    for (int it = 0; it < NT; ++it)
      acc[it] = __builtin_amdgcn_wmma_f32_16x16x32_f16(
          false, a, false, bf[it], (short)0, acc[it], false, false);
  }

  // ---- fused first-spike detection ----------------------------------------
  // C layout: lane<16 -> M=r, N=lane ; lane>=16 -> M=r+8, N=lane-16
  int mOff  = laneHi << 3;
  int nLane = lane & 15;
#pragma unroll
  for (int r = 0; r < 8; ++r) {
    int row = mTile * 16 + mOff + r;
    int b   = row / kNumT;
    int t   = row - b * kNumT;
#pragma unroll
    for (int it = 0; it < NT; ++it) {
      if (acc[it][r] >= 1.0f) {
        int n = (ngBase + it) * 16 + nLane;
        atomicMin(&idx[(size_t)b * idxStride + n], (unsigned)t);
      }
    }
  }
}

// ---------------------------------------------------------------------------
// Host launcher
// ---------------------------------------------------------------------------
extern "C" void kernel_launch(void* const* d_in, const int* in_sizes, int n_in,
                              void* d_out, int out_size, void* d_ws, size_t ws_size,
                              hipStream_t stream) {
  const float* ti = (const float*)d_in[0];   // [64, 1876]
  const float* W1 = (const float*)d_in[1];   // [1024, 1876]
  const float* W2 = (const float*)d_in[2];   // [10, 1024]
  float* out = (float*)d_out;                // [64, 10]

  constexpr int Bsz = 64, F0 = 1876, F1 = 1024, F2 = 10;
  constexpr int M      = Bsz * kNumT;        // 9664 (= 604 * 16)
  constexpr int mTiles = M / 16;             // 604
  constexpr int Kpad1  = 1888;               // 1876 -> mult of 32
  constexpr int Ntiles1 = F1 / 16;           // 64
  constexpr int NT1 = 4, nGroups1 = Ntiles1 / NT1;  // 16
  constexpr int Kpad2  = 1024;
  constexpr int Ntiles2 = 1;                 // N=10 -> one padded 16-col tile

  // Workspace carve-up (all chunks 256B-aligned by construction; ~4.9 MB)
  char* ws = (char*)d_ws;
  size_t off = 0;
  float*    tiP1 = (float*)(ws + off);    off += (size_t)Bsz * Kpad1 * 4;
  _Float16* B1p  = (_Float16*)(ws + off); off += (size_t)(Kpad1 / 32) * Ntiles1 * 512 * 2;
  unsigned* idx1 = (unsigned*)(ws + off); off += (size_t)Bsz * F1 * 4;
  float*    tiP2 = (float*)(ws + off);    off += (size_t)Bsz * F1 * 4;
  _Float16* B2p  = (_Float16*)(ws + off); off += (size_t)(Kpad2 / 32) * Ntiles2 * 512 * 2;
  unsigned* idx2 = (unsigned*)(ws + off); off += (size_t)Bsz * 16 * 4;

  auto nb = [](int n) { return (n + 255) / 256; };

  // Prep
  { int tot = Bsz * Kpad1;
    pad_ti_kernel<<<nb(tot), 256, 0, stream>>>(ti, tiP1, F0, Kpad1, tot); }
  { int tot = (Kpad1 / 32) * Ntiles1 * 512;
    pack_w_kernel<<<nb(tot), 256, 0, stream>>>(W1, B1p, F0, F1, Ntiles1, tot); }
  { int tot = Bsz * F1;
    fill_u32_kernel<<<nb(tot), 256, 0, stream>>>(idx1, kLastT, tot); }
  { int tot = Bsz * 16;
    fill_u32_kernel<<<nb(tot), 256, 0, stream>>>(idx2, kLastT, tot); }

  // Layer 1: [9664 x 1888] x [1888 x 1024] fused with first-spike atomicMin
  { int waves = mTiles * nGroups1;           // 9664 waves
    int thr = waves * 32;
    spike_gemm_kernel<NT1><<<nb(thr), 256, 0, stream>>>(
        tiP1, Kpad1, B1p, Ntiles1, idx1, F1, nGroups1, waves); }

  // h1 = idx1 * dt  (layer-2 spike-time input)
  { int tot = Bsz * F1;
    idx_to_time_kernel<<<nb(tot), 256, 0, stream>>>(idx1, tiP2, tot); }
  { int tot = (Kpad2 / 32) * Ntiles2 * 512;
    pack_w_kernel<<<nb(tot), 256, 0, stream>>>(W2, B2p, F1, F2, Ntiles2, tot); }

  // Layer 2: [9664 x 1024] x [1024 x 16(pad of 10)]
  { int waves = mTiles;                      // 604 waves
    int thr = waves * 32;
    spike_gemm_kernel<1><<<nb(thr), 256, 0, stream>>>(
        tiP2, Kpad2, B2p, Ntiles2, idx2, 16, 1, waves); }

  // out = idx2 * dt (first 10 cols)
  { int tot = Bsz * F2;
    final_out_kernel<<<nb(tot), 256, 0, stream>>>(idx2, out, tot); }

  (void)in_sizes; (void)n_in; (void)out_size; (void)ws_size; (void)off;
}
